// LSTM_20864951124250
// MI455X (gfx1250) — matrix operand order repaired and checked by reading, not compiled, not run on hardware
//
#include <hip/hip_runtime.h>
#include <hip/hip_bf16.h>

#define FEAT 8
#define HID 16
#define TSTEPS 4
#define TILE_M 16
#define WAVES_PER_BLOCK 8

typedef __attribute__((ext_vector_type(16))) _Float16 v16h;
typedef __attribute__((ext_vector_type(8)))  _Float16 v8h;
typedef __attribute__((ext_vector_type(8)))  float    v8f;

// gfx1250 has a hardware tanh transcendental (v_tanh_f32): 1 trans op instead
// of exp+rcp (2 trans ops). Fall back to exp2/rcp if the builtin is absent.
__device__ __forceinline__ float dev_tanh(float x) {
#if __has_builtin(__builtin_amdgcn_tanhf)
    return __builtin_amdgcn_tanhf(x);
#else
    return 2.0f * __builtin_amdgcn_rcpf(1.0f + __builtin_amdgcn_exp2f(-2.88539008f * x)) - 1.0f;
#endif
}
__device__ __forceinline__ float dev_sigmoid(float x) {
    // sigmoid(x) = 0.5*tanh(x/2) + 0.5  -> 1 trans + fma
    return __builtin_fmaf(0.5f, dev_tanh(0.5f * x), 0.5f);
}

// Compiler-only ordering fence. Hardware DS ops are in-order within a wave
// (all LDS tiles here are per-wave private), so no s_barrier is needed; we
// only need to stop LLVM reordering LDS loads past LDS stores.
__device__ __forceinline__ void lds_fence() { asm volatile("" ::: "memory"); }

// 16-byte LDS row load without TBAA no-alias assumptions.
__device__ __forceinline__ v8h load_row8(const _Float16* p) {
    v8h v;
    __builtin_memcpy(&v, p, sizeof(v8h));
    return v;
}

// One wave32 processes TILE_M=16 batch rows through the full 2-layer LSTM + FC.
// WMMA A layout (16-bit 16x32): lane<16 holds row M=lane, K 0..7 (elems 0..7) and
//   K 16..23 (elems 8..15); lane>=16 holds row M=lane-16, K 8..15 and K 24..31.
// WMMA B layout (16-bit 32x16): lane<16 holds col N=lane, K 0..7 and K 16..23; lane>=16 K 8..15 and K 24..31.
// C/D layout (f32 16x16): lane holds col N=lane&15, rows M = (lane>>4)*8 + r in VGPR r.
__global__ void __launch_bounds__(256, 1) lstm2_wmma_kernel(
    const float* __restrict__ x,
    const float* __restrict__ w_ih0, const float* __restrict__ w_hh0,
    const float* __restrict__ b_ih0, const float* __restrict__ b_hh0,
    const float* __restrict__ w_ih1, const float* __restrict__ w_hh1,
    const float* __restrict__ b_ih1, const float* __restrict__ b_hh1,
    const float* __restrict__ fc_w,  const float* __restrict__ fc_b,
    float* __restrict__ out)
{
    // per-wave h staging tiles: [wave][layer][M=16][N=16] f16 (strictly wave-private)
    __shared__ __align__(32) _Float16 sH[WAVES_PER_BLOCK * 2 * TILE_M * HID];

    const int tid  = threadIdx.x;
    const int lane = tid & 31;
    const int wave = tid >> 5;
    const int col  = lane & 15;   // N index (gate/hidden) for B/C; batch row for A
    const int hi   = lane >> 4;   // which half of the wave

    const long base = ((long)blockIdx.x * WAVES_PER_BLOCK + wave) * TILE_M;

    _Float16* H0 = &sH[(wave * 2 + 0) * TILE_M * HID]; // layer0 h (also layer1 input)
    _Float16* H1 = &sH[(wave * 2 + 1) * TILE_M * HID]; // layer1 h

    // h(-1) = 0
    #pragma unroll
    for (int r = 0; r < 8; ++r) {
        const int m = hi * 8 + r;
        H0[m * HID + col] = (_Float16)0.0f;
        H1[m * HID + col] = (_Float16)0.0f;
    }
    lds_fence();

    // ---- Build resident B fragments: B = [W_ih^T ; W_hh^T] stacked along K ----
    // layer0: K 0..7 = W_ih0 (F=8), K 8..23 = W_hh0, K 24..31 = 0 (pad)
    // layer1: K 0..15 = W_ih1, K 16..31 = W_hh1
    v16h B0[4], B1[4];
    float bias0[4], bias1[4];
    #pragma unroll
    for (int g = 0; g < 4; ++g) {
        const int n = g * 16 + col;
        #pragma unroll
        for (int j = 0; j < 16; ++j) {
            const int K = (j < 8) ? hi * 8 + j : 16 + hi * 8 + (j - 8);
            float w0 = (K < FEAT)        ? w_ih0[n * FEAT + K]
                     : (K < FEAT + HID)  ? w_hh0[n * HID + (K - FEAT)]
                                         : 0.0f;
            B0[g][j] = (_Float16)w0;
            float w1 = (K < HID) ? w_ih1[n * HID + K] : w_hh1[n * HID + (K - HID)];
            B1[g][j] = (_Float16)w1;
        }
        bias0[g] = b_ih0[n] + b_hh0[n];
        bias1[g] = b_ih1[n] + b_hh1[n];
    }

    v8f c0, c1;
    #pragma unroll
    for (int r = 0; r < 8; ++r) { c0[r] = 0.0f; c1[r] = 0.0f; }

    #pragma unroll
    for (int t = 0; t < TSTEPS; ++t) {
        // ================= layer 0: A = [x_t | h0] =================
        v16h a;
        if (hi == 0) {
            const float* xp = x + (base + col) * (TSTEPS * FEAT) + t * FEAT;
            #pragma unroll
            for (int j = 0; j < 8; ++j) a[j] = (_Float16)xp[j];       // K=0..7
            v8h hv = load_row8(H0 + col * HID + 8);                   // K=16..23 -> h[8..15]
            #pragma unroll
            for (int j = 0; j < 8; ++j) a[8 + j] = hv[j];
        } else {
            v8h hv = load_row8(H0 + col * HID + 0);                   // K=8..15 -> h[0..7]
            #pragma unroll
            for (int j = 0; j < 8; ++j) a[j] = hv[j];
            #pragma unroll
            for (int j = 0; j < 8; ++j) a[8 + j] = (_Float16)0.0f;    // K=24..31 pad
        }

        v8f gates[4];
        #pragma unroll
        for (int g = 0; g < 4; ++g) {
            v8f cacc;
            #pragma unroll
            for (int r = 0; r < 8; ++r) cacc[r] = bias0[g];
            gates[g] = __builtin_amdgcn_wmma_f32_16x16x32_f16(
                false, a, false, B0[g], (short)0, cacc, false, false);
        }
        lds_fence();  // all H0 reads for this step precede the overwrite below

        #pragma unroll
        for (int r = 0; r < 8; ++r) {
            const float ig = dev_sigmoid(gates[0][r]);
            const float fg = dev_sigmoid(gates[1][r]);
            const float gv = dev_tanh   (gates[2][r]);
            const float og = dev_sigmoid(gates[3][r]);
            const float cv = fg * c0[r] + ig * gv;
            c0[r] = cv;
            const float hv = og * dev_tanh(cv);
            H0[(hi * 8 + r) * HID + col] = (_Float16)hv;  // transpose via LDS
        }
        lds_fence();

        // ================= layer 1: A = [h0_t | h1] =================
        {
            v8h hin = load_row8(H0 + col * HID + hi * 8);  // K=0..15
            v8h hst = load_row8(H1 + col * HID + hi * 8);  // K=16..31
            #pragma unroll
            for (int j = 0; j < 8; ++j) { a[j] = hin[j]; a[8 + j] = hst[j]; }
        }
        #pragma unroll
        for (int g = 0; g < 4; ++g) {
            v8f cacc;
            #pragma unroll
            for (int r = 0; r < 8; ++r) cacc[r] = bias1[g];
            gates[g] = __builtin_amdgcn_wmma_f32_16x16x32_f16(
                false, a, false, B1[g], (short)0, cacc, false, false);
        }
        lds_fence();

        #pragma unroll
        for (int r = 0; r < 8; ++r) {
            const float ig = dev_sigmoid(gates[0][r]);
            const float fg = dev_sigmoid(gates[1][r]);
            const float gv = dev_tanh   (gates[2][r]);
            const float og = dev_sigmoid(gates[3][r]);
            const float cv = fg * c1[r] + ig * gv;
            c1[r] = cv;
            const float hv = og * dev_tanh(cv);
            H1[(hi * 8 + r) * HID + col] = (_Float16)hv;
        }
        lds_fence();
    }

    // ================= FC on final h2 =================
    if (hi == 0) {
        const _Float16* row = H1 + col * HID;   // batch row M = lane
        float acc = fc_b[0];
        #pragma unroll
        for (int n = 0; n < HID; ++n) acc += (float)row[n] * fc_w[n];
        out[base + col] = acc;
    }
}

extern "C" void kernel_launch(void* const* d_in, const int* in_sizes, int n_in,
                              void* d_out, int out_size, void* d_ws, size_t ws_size,
                              hipStream_t stream) {
    const float* x     = (const float*)d_in[0];
    const float* w_ih0 = (const float*)d_in[1];
    const float* w_hh0 = (const float*)d_in[2];
    const float* b_ih0 = (const float*)d_in[3];
    const float* b_hh0 = (const float*)d_in[4];
    const float* w_ih1 = (const float*)d_in[5];
    const float* w_hh1 = (const float*)d_in[6];
    const float* b_ih1 = (const float*)d_in[7];
    const float* b_hh1 = (const float*)d_in[8];
    const float* fc_w  = (const float*)d_in[9];
    const float* fc_b  = (const float*)d_in[10];
    float* out = (float*)d_out;

    const int batch = in_sizes[0] / (TSTEPS * FEAT);          // 1048576
    const int blocks = batch / (TILE_M * WAVES_PER_BLOCK);    // 8192

    lstm2_wmma_kernel<<<blocks, 32 * WAVES_PER_BLOCK, 0, stream>>>(
        x, w_ih0, w_hh0, b_ih0, b_hh0, w_ih1, w_hh1, b_ih1, b_hh1, fc_w, fc_b, out);
}
